// EfficientMemoryGEMMWithSoftmax_39900246180593
// MI455X (gfx1250) — matrix operand order repaired and checked by reading, not compiled, not run on hardware
//
#include <hip/hip_runtime.h>
#include <hip/hip_bf16.h>

// ---------------------------------------------------------------------------
// Problem constants (from reference): B=2, H=16, S=2048, D=128
// ---------------------------------------------------------------------------
#define S_DIM 2048
#define D_DIM 128
#define BH    32                      // B*H
#define RESULT_ELEMS (2L * 16 * 2048 * 128)   // 8,388,608
#define OFF_O1    (RESULT_ELEMS + 0)
#define OFF_O2    (RESULT_ELEMS + 1)
#define OFF_COLS  (RESULT_ELEMS + 2)
#define N_COLS    25                  // int(0.2 * 128)
#define OFF_SCALE (OFF_COLS + N_COLS)

typedef __attribute__((ext_vector_type(8)))  float          v8f;
typedef __attribute__((ext_vector_type(16))) __bf16         v16bf;
typedef __attribute__((ext_vector_type(16))) unsigned short v16s;
typedef __attribute__((ext_vector_type(8)))  unsigned short v8s;
typedef __attribute__((ext_vector_type(4)))  unsigned short v4s;

__device__ __forceinline__ unsigned short f2bf(float f) {
    // round-to-nearest-even fp32 -> bf16
    unsigned u = __float_as_uint(f);
    unsigned r = u + 0x7FFFu + ((u >> 16) & 1u);
    return (unsigned short)(r >> 16);
}

// ---------------------------------------------------------------------------
// Batched GEMM: C[bh] = A[bh](2048x2048) * B[bh](2048x128), bf16 WMMA, f32 acc
// Block: 256 threads (8 waves). Each block: 128 rows x 128 cols output tile.
// Each wave: 16 rows x 128 cols = 8 accumulators of 16x16.
// ---------------------------------------------------------------------------
__global__ __launch_bounds__(256)
void gemm_wmma_bf16(const float* __restrict__ X1, const float* __restrict__ X2,
                    float* __restrict__ Out) {
    __shared__ alignas(16) unsigned short ldsA[128 * 32];   // [row][k] bf16
    __shared__ alignas(16) unsigned short ldsB[128 * 32];   // [n][k]  bf16 (transposed)

    const int bh = blockIdx.y;
    const int m0 = blockIdx.x * 128;
    const float* A  = X1  + (size_t)bh * S_DIM * S_DIM;
    const float* Bm = X2  + (size_t)bh * S_DIM * D_DIM;
    float*       C  = Out + (size_t)bh * S_DIM * D_DIM;

    const int tid  = threadIdx.x;
    const int lane = tid & 31;
    const int w    = tid >> 5;        // wave id 0..7
    const int g    = lane >> 4;       // lane group (0: lanes 0-15, 1: lanes 16-31)
    const int l15  = lane & 15;

    v8f acc[8];
#pragma unroll
    for (int t = 0; t < 8; ++t) acc[t] = (v8f)0.0f;

    for (int k0 = 0; k0 < S_DIM; k0 += 32) {
        // ---- stage A tile (128 rows x 32 k), fp32 -> bf16 ----
#pragma unroll
        for (int i = 0; i < 4; ++i) {
            const int idx = tid + 256 * i;          // 0..1023
            const int r = idx >> 3;                 // row 0..127
            const int q = idx & 7;                  // k-quad 0..7
            const float4 v = *(const float4*)(A + (size_t)(m0 + r) * S_DIM + k0 + 4 * q);
            v4s h = { f2bf(v.x), f2bf(v.y), f2bf(v.z), f2bf(v.w) };
            *(v4s*)&ldsA[r * 32 + 4 * q] = h;
            if (k0 + 32 < S_DIM)   // prefetch next A k-tile (global_prefetch_b8)
                __builtin_prefetch(A + (size_t)(m0 + r) * S_DIM + k0 + 32 + 4 * q, 0, 0);
        }
        // ---- stage B tile (32 k x 128 n) transposed into LDS [n][k] ----
#pragma unroll
        for (int i = 0; i < 4; ++i) {
            const int idx = tid + 256 * i;          // 0..1023
            const int K  = idx & 31;                // k 0..31
            const int nq = idx >> 5;                // n-quad 0..31
            const float4 v = *(const float4*)(Bm + (size_t)(k0 + K) * D_DIM + 4 * nq);
            ldsB[(4 * nq + 0) * 32 + K] = f2bf(v.x);
            ldsB[(4 * nq + 1) * 32 + K] = f2bf(v.y);
            ldsB[(4 * nq + 2) * 32 + K] = f2bf(v.z);
            ldsB[(4 * nq + 3) * 32 + K] = f2bf(v.w);
        }
        __syncthreads();

        // ---- A fragment: 16x32 bf16, contiguous k order (same as B):
        // lane (g,l15) = row l15; elem e -> K=16g+e
        const int rloc = 16 * w + l15;
        v8s a0 = *(const v8s*)&ldsA[rloc * 32 + 16 * g];
        v8s a1 = *(const v8s*)&ldsA[rloc * 32 + 16 * g + 8];
        v16s av = __builtin_shufflevector(a0, a1, 0,1,2,3,4,5,6,7,8,9,10,11,12,13,14,15);
        v16bf afrag = __builtin_bit_cast(v16bf, av);

        // ---- 8 WMMA per K-step: B fragment 32x16 per n-tile
        // lane (g,l15): col = 16t+l15; elem e -> K = 16g+e (contiguous in LDS [n][k])
#pragma unroll
        for (int t = 0; t < 8; ++t) {
            const unsigned short* pb = &ldsB[(16 * t + l15) * 32 + 16 * g];
            v8s b0 = *(const v8s*)pb;
            v8s b1 = *(const v8s*)(pb + 8);
            v16s bv = __builtin_shufflevector(b0, b1, 0,1,2,3,4,5,6,7,8,9,10,11,12,13,14,15);
            v16bf bfrag = __builtin_bit_cast(v16bf, bv);
            acc[t] = __builtin_amdgcn_wmma_f32_16x16x32_bf16(
                false, afrag, false, bfrag, (short)0, acc[t], false, false);
        }
        __syncthreads();
    }

    // ---- store: C/D layout: VGPR r, lane(g,l15): M = r + 8g, N = l15 ----
    float* cp = C + (size_t)(m0 + 16 * w + 8 * g) * D_DIM + l15;
#pragma unroll
    for (int t = 0; t < 8; ++t)
#pragma unroll
        for (int r = 0; r < 8; ++r)
            cp[(size_t)r * D_DIM + 16 * t] = acc[t][r];
}

// ---------------------------------------------------------------------------
// Exact quantile via IEEE-bit radix histogram (values are non-negative:
// x1 is softmax output; x2 path uses |x|). 2 passes + 2 tiny scans.
// ---------------------------------------------------------------------------
__global__ void zero_u32(unsigned* __restrict__ p, int n) {
    int i = blockIdx.x * 256 + threadIdx.x;
    if (i < n) p[i] = 0u;
}

__global__ __launch_bounds__(256)
void hist_coarse(const float* __restrict__ x, long n, unsigned* __restrict__ h) {
    __shared__ unsigned sh[16384];
    for (int i = threadIdx.x; i < 16384; i += 256) sh[i] = 0u;
    __syncthreads();
    const long stride = (long)gridDim.x * 256;
    for (long i = (long)blockIdx.x * 256 + threadIdx.x; i < n; i += stride) {
        unsigned b = __float_as_uint(x[i]) & 0x7FFFFFFFu;
        atomicAdd(&sh[b >> 17], 1u);
    }
    __syncthreads();
    for (int i = threadIdx.x; i < 16384; i += 256)
        if (sh[i]) atomicAdd(&h[i], sh[i]);
}

__global__ void scan_coarse(const unsigned* __restrict__ h, long rank,
                            unsigned* __restrict__ out_bin, unsigned* __restrict__ out_rank) {
    unsigned long cum = 0;
    for (int b = 0; b < 16384; ++b) {
        unsigned c = h[b];
        if (cum + c > (unsigned long)rank) { *out_bin = (unsigned)b; *out_rank = (unsigned)(rank - cum); return; }
        cum += c;
    }
    *out_bin = 16383u; *out_rank = 0u;
}

__global__ __launch_bounds__(256)
void hist_fine(const float* __restrict__ x, long n,
               const unsigned* __restrict__ binp, unsigned* __restrict__ h2) {
    const unsigned bin = *binp;
    const long stride = (long)gridDim.x * 256;
    for (long i = (long)blockIdx.x * 256 + threadIdx.x; i < n; i += stride) {
        unsigned b = __float_as_uint(x[i]) & 0x7FFFFFFFu;
        if ((b >> 17) == bin) atomicAdd(&h2[b & 0x1FFFFu], 1u);
    }
}

__global__ void scan_fine(const unsigned* __restrict__ h2, const unsigned* __restrict__ binp,
                          const unsigned* __restrict__ rankp, float* __restrict__ out_val) {
    const unsigned bin = *binp, rank = *rankp;
    unsigned long cum = 0;
    for (unsigned b = 0; b < 131072u; ++b) {
        unsigned c = h2[b];
        if (cum + c > (unsigned long)rank) { *out_val = __uint_as_float((bin << 17) | b); return; }
        cum += c;
    }
    *out_val = __uint_as_float((bin << 17) | 131071u);
}

// ---------------------------------------------------------------------------
// Per-column sum-of-squares and max-abs of residual (|x| <= outlier kept)
// over x2[0] (32768 rows x 128 cols).
// ---------------------------------------------------------------------------
__global__ __launch_bounds__(256)
void colstats(const float* __restrict__ x, long n, const float* __restrict__ outlp,
              float* __restrict__ gsq, unsigned* __restrict__ gmax) {
    __shared__ float    ssq[128];
    __shared__ unsigned smax[128];
    if (threadIdx.x < 128) { ssq[threadIdx.x] = 0.0f; smax[threadIdx.x] = 0u; }
    __syncthreads();
    const float outl = *outlp;
    const long stride = (long)gridDim.x * 256;
    for (long i = (long)blockIdx.x * 256 + threadIdx.x; i < n; i += stride) {
        float v = x[i];
        float a = fabsf(v);
        if (a <= outl) {                       // where(|x| > outl, 0, x)
            int c = (int)(i & 127);
            atomicAdd(&ssq[c], v * v);
            atomicMax(&smax[c], __float_as_uint(a));   // bit order == value order (a>=0)
        }
    }
    __syncthreads();
    if (threadIdx.x < 128) {
        atomicAdd(&gsq[threadIdx.x], ssq[threadIdx.x]);
        atomicMax(&gmax[threadIdx.x], smax[threadIdx.x]);
    }
}

// top-25 columns (descending, like lax.top_k), scale, and scalar outputs
__global__ void finalize(const float* __restrict__ gsq, const unsigned* __restrict__ gmax,
                         const float* __restrict__ o1p, const float* __restrict__ o2p,
                         float* __restrict__ out) {
    out[OFF_O1] = *o1p;
    out[OFF_O2] = *o2p;
    bool used[128];
    for (int c = 0; c < 128; ++c) used[c] = false;
    float m = 0.0f;
    for (int j = 0; j < N_COLS; ++j) {
        int best = 0; float bv = -1.0f;
        for (int c = 0; c < 128; ++c)
            if (!used[c] && gsq[c] > bv) { bv = gsq[c]; best = c; }
        used[best] = true;
        out[OFF_COLS + j] = (float)best;
        float cm = __uint_as_float(gmax[best]);
        if (cm > m) m = cm;
    }
    out[OFF_SCALE] = m / 127.0f;
}

// ---------------------------------------------------------------------------
// Launch
// ---------------------------------------------------------------------------
extern "C" void kernel_launch(void* const* d_in, const int* in_sizes, int n_in,
                              void* d_out, int out_size, void* d_ws, size_t ws_size,
                              hipStream_t stream) {
    const float* x1 = (const float*)d_in[0];
    const float* x2 = (const float*)d_in[1];
    float* out = (float*)d_out;
    unsigned* ws = (unsigned*)d_ws;

    // workspace layout (uint32 units)
    unsigned* h1a  = ws;                 // 16384
    unsigned* h2a  = ws + 16384;         // 131072
    unsigned* h1b  = ws + 147456;        // 16384
    unsigned* h2b  = ws + 163840;        // 131072
    float*    gsq  = (float*)(ws + 294912);   // 128
    unsigned* gmax = ws + 295040;        // 128
    unsigned* scal = ws + 295168;        // 8 scalars
    unsigned* b1a = scal + 0; unsigned* r1a = scal + 1; float* o1 = (float*)(scal + 2);
    unsigned* b1b = scal + 3; unsigned* r1b = scal + 4; float* o2 = (float*)(scal + 5);
    const int ws_u32 = 295176;

    // 0) zero workspace
    zero_u32<<<(ws_u32 + 255) / 256, 256, 0, stream>>>(ws, ws_u32);

    // 1) batched GEMM (dominant cost; HBM-bound ~25us floor)
    gemm_wmma_bf16<<<dim3(S_DIM / 128, BH), 256, 0, stream>>>(x1, x2, out);

    // 2) outliner_1: exact 99th percentile of x1[0] (H*S*S elements)
    const long N1 = 16L * S_DIM * S_DIM;
    const long rank1 = (long)((double)N1 * 0.99) - 1;   // sorted[k-1]
    hist_coarse<<<2048, 256, 0, stream>>>(x1, N1, h1a);
    scan_coarse<<<1, 1, 0, stream>>>(h1a, rank1, b1a, r1a);
    hist_fine<<<2048, 256, 0, stream>>>(x1, N1, b1a, h2a);
    scan_fine<<<1, 1, 0, stream>>>(h2a, b1a, r1a, o1);

    // 3) outliner_2: exact 99th percentile of |x2[0]| (H*S*D elements)
    const long N2 = 16L * S_DIM * D_DIM;
    const long rank2 = (long)((double)N2 * 0.99) - 1;
    hist_coarse<<<256, 256, 0, stream>>>(x2, N2, h1b);
    scan_coarse<<<1, 1, 0, stream>>>(h1b, rank2, b1b, r1b);
    hist_fine<<<256, 256, 0, stream>>>(x2, N2, b1b, h2b);
    scan_fine<<<1, 1, 0, stream>>>(h2b, b1b, r1b, o2);

    // 4) column norms / max-abs of residual, then top-25 cols + scale
    colstats<<<512, 256, 0, stream>>>(x2, N2, o2, gsq, gmax);
    finalize<<<1, 1, 0, stream>>>(gsq, gmax, o1, o2, out);
}
